// mBP_model_47802986004493
// MI455X (gfx1250) — matrix operand order repaired and checked by reading, not compiled, not run on hardware
//
#include <hip/hip_runtime.h>
#include <hip/hip_bf16.h>

typedef __attribute__((ext_vector_type(16))) _Float16 v16h;
typedef __attribute__((ext_vector_type(8)))  _Float16 v8h;
typedef __attribute__((ext_vector_type(8)))  float    v8f;

#define NAT    10000
#define NN     32
#define RSN    16
#define THETAN 8
#define NEMB   4
#define PBOND  64      // RSN*NEMB
#define FEAT   576     // PBOND + THETAN*PBOND
#define HID    64
#define RC_    5.0f
#define PI_    3.14159265358979323846f

union V16H { v16h v; v8h h8[2]; _Float16 h[16]; };

__device__ inline v16h load_a_op(const _Float16* p0, const _Float16* p1) {
    V16H t;
    t.h8[0] = *(const v8h*)p0;   // halves 0..7
    t.h8[1] = *(const v8h*)p1;   // halves 8..15
    return t.v;
}

// ---------------------------------------------------------------------------
// Kernel A: per-atom descriptor (radial + WMMA angular). One wave32 per atom.
// ---------------------------------------------------------------------------
__global__ __launch_bounds__(32) void desc_kernel(
    const float* __restrict__ pos, const float* __restrict__ spe,
    const float* __restrict__ theta_s, const float* __restrict__ kn_rad,
    const int* __restrict__ nbr, float* __restrict__ desc)
{
    __shared__ float    u[NN][4];
    __shared__ float    cosm[NN][NN];
    __shared__ float    sinm[NN][NN];
    __shared__ _Float16 A16[NN][NN];      // A16[j][k] per theta
    __shared__ _Float16 Pt[PBOND][NN];    // Pt[p][j] = P[j][p]  (f16)

    const int atom = blockIdx.x;
    const int lane = threadIdx.x;         // lane == neighbor index j
    const int half = lane >> 4;           // 0: lanes 0-15, 1: lanes 16-31
    const int l15  = lane & 15;

    const int nb = nbr[atom * NN + lane];

    const float px = pos[atom * 3 + 0], py = pos[atom * 3 + 1], pz = pos[atom * 3 + 2];
    const float dx = pos[nb * 3 + 0] - px;
    const float dy = pos[nb * 3 + 1] - py;
    const float dz = pos[nb * 3 + 2] - pz;
    const float rn = sqrtf(dx * dx + dy * dy + dz * dz) + 1e-8f;
    const float inv_rn = 1.0f / rn;

    u[lane][0] = dx * inv_rn;
    u[lane][1] = dy * inv_rn;
    u[lane][2] = dz * inv_rn;
    u[lane][3] = 0.0f;

    // cosine cutoff
    const float fc = (rn < RC_) ? 0.5f * (cosf(PI_ * rn / RC_) + 1.0f) : 0.0f;

    // species embedding product (center * neighbor)
    float semb[NEMB];
    #pragma unroll
    for (int m = 0; m < NEMB; ++m)
        semb[m] = spe[nb * NEMB + m] * spe[atom * NEMB + m];

    // bessel radial basis and per-bond channel tensor P -> Pt (f16, transposed)
    const float kn    = kn_rad[0];
    const float bpref = sqrtf(2.0f / RC_);
    #pragma unroll
    for (int l = 0; l < RSN; ++l) {
        const float arg = (PI_ / RC_) * kn * (float)(l + 1) * rn;
        const float bf  = bpref * sinf(arg) * inv_rn * fc;
        #pragma unroll
        for (int m = 0; m < NEMB; ++m)
            Pt[l * NEMB + m][lane] = (_Float16)(bf * semb[m]);
    }
    __syncthreads();

    // ---- radial descriptor: desc[atom, p] = sum_j P[j][p]  (p = 0..63) ----
    {
        float s0 = 0.0f, s1 = 0.0f;
        #pragma unroll
        for (int j = 0; j < NN; ++j) {
            s0 += (float)Pt[lane][j];
            s1 += (float)Pt[lane + 32][j];
        }
        desc[atom * FEAT + lane]      = s0;
        desc[atom * FEAT + lane + 32] = s1;
    }

    // ---- pairwise cos/sin matrices (row per lane) ----
    {
        const float ux = u[lane][0], uy = u[lane][1], uz = u[lane][2];
        #pragma unroll 4
        for (int k = 0; k < NN; ++k) {
            float c = ux * u[k][0] + uy * u[k][1] + uz * u[k][2];
            c = fminf(fmaxf(c, -1.0f + 1e-6f), 1.0f - 1e-6f);
            cosm[lane][k] = c;
            sinm[lane][k] = sqrtf(1.0f - c * c);
        }
    }
    __syncthreads();

    const float apref = 0x1p-15f;   // 2^(1 - ZETA), ZETA = 16

    for (int t = 0; t < THETAN; ++t) {
        const float th = theta_s[t];
        const float ct = cosf(th), st = sinf(th);

        // A_t[j][k] = 2^-15 * (1 + cos(theta_jk - theta_t))^16   (f16)
        #pragma unroll 4
        for (int k = 0; k < NN; ++k) {
            const float cd = cosm[lane][k] * ct + sinm[lane][k] * st;
            const float b  = 1.0f + cd;
            const float b2 = b * b, b4 = b2 * b2, b8 = b4 * b4;
            A16[lane][k] = (_Float16)(apref * (b8 * b8));
        }
        __syncthreads();

        // C = A (32x32) * P (32x64); ang[p] = sum_k C[k,p]*P[k,p]
        #pragma unroll
        for (int pt = 0; pt < 4; ++pt) {
            const int p = pt * 16 + l15;
            // B operand: column p, K = half*8 .. +7 and 16+half*8 .. +7 (table order)
            const v16h bop = load_a_op(&Pt[p][half * 8], &Pt[p][half * 8 + 16]);

            float partial = 0.0f;
            #pragma unroll
            for (int kt = 0; kt < 2; ++kt) {
                const int arow  = kt * 16 + l15;
                const int abase = half * 8;
                const v16h aop = load_a_op(&A16[arow][abase], &A16[arow][abase + 16]);
                v8f acc = {};
                acc = __builtin_amdgcn_wmma_f32_16x16x32_f16(
                    false, aop, false, bop, (short)0, acc, false, false);
                // elementwise with P in C-layout: k = kt*16 + half*8 + r
                const v8h pk = *(const v8h*)&Pt[p][kt * 16 + half * 8];
                #pragma unroll
                for (int r = 0; r < 8; ++r)
                    partial += acc[r] * (float)pk[r];
            }
            partial += __shfl_xor(partial, 16, 32);
            if (lane < 16)
                desc[atom * FEAT + PBOND + t * PBOND + pt * 16 + lane] = partial;
        }
        __syncthreads();   // protect A16 rewrite next theta
    }
}

// ---------------------------------------------------------------------------
// Prep: zero the output scalar; transpose+convert W1,W2 to f16 (N-major rows)
// ---------------------------------------------------------------------------
__global__ void prep_kernel(const float* __restrict__ W1, const float* __restrict__ W2,
                            _Float16* __restrict__ W1t, _Float16* __restrict__ W2t,
                            float* __restrict__ out)
{
    const int i = blockIdx.x * blockDim.x + threadIdx.x;
    if (i == 0) out[0] = 0.0f;
    if (i < FEAT * HID) {              // W1t[n*576 + k] = W1[k*64 + n]
        const int n = i / FEAT, k = i % FEAT;
        W1t[i] = (_Float16)W1[k * HID + n];
    }
    if (i < HID * HID) {               // W2t[n*64 + k] = W2[k*64 + n]
        const int n = i / HID, k = i % HID;
        W2t[i] = (_Float16)W2[k * HID + n];
    }
}

// ---------------------------------------------------------------------------
// Kernel B: MLP 576->64 tanh -> 64 tanh -> 1, summed into out[0].
// One wave32 per 16-atom tile (NAT = 625 * 16 exactly).
// ---------------------------------------------------------------------------
__global__ __launch_bounds__(32) void mlp_kernel(
    const float* __restrict__ desc, const _Float16* __restrict__ W1t,
    const float* __restrict__ b1,  const _Float16* __restrict__ W2t,
    const float* __restrict__ b2,  const float* __restrict__ W3,
    const float* __restrict__ b3,  float* __restrict__ out)
{
    __shared__ _Float16 h1[16][HID];
    __shared__ float    h2[16][HID];

    const int tile = blockIdx.x;
    const int lane = threadIdx.x;
    const int half = lane >> 4;
    const int l15  = lane & 15;
    const int row  = tile * 16 + l15;   // A-operand row (atom index)

    const v8f vz = {0.f, 0.f, 0.f, 0.f, 0.f, 0.f, 0.f, 0.f};
    v8f acc[4] = {vz, vz, vz, vz};

    // ---- GEMM1: desc[16 x 576] @ W1[576 x 64] ----
    for (int kb = 0; kb < FEAT / 32; ++kb) {
        const int c0 = kb * 32 + half * 8;
        const float4* q0 = (const float4*)(desc + row * FEAT + c0);
        const float4* q1 = (const float4*)(desc + row * FEAT + c0 + 16);
        const float4 f0 = q0[0], f1 = q0[1];   // K = c0 .. c0+7
        const float4 f2 = q1[0], f3 = q1[1];   // K = c0+16 .. c0+23
        V16H a;
        a.h[0]  = (_Float16)f0.x; a.h[1]  = (_Float16)f0.y;
        a.h[2]  = (_Float16)f0.z; a.h[3]  = (_Float16)f0.w;
        a.h[4]  = (_Float16)f1.x; a.h[5]  = (_Float16)f1.y;
        a.h[6]  = (_Float16)f1.z; a.h[7]  = (_Float16)f1.w;
        a.h[8]  = (_Float16)f2.x; a.h[9]  = (_Float16)f2.y;
        a.h[10] = (_Float16)f2.z; a.h[11] = (_Float16)f2.w;
        a.h[12] = (_Float16)f3.x; a.h[13] = (_Float16)f3.y;
        a.h[14] = (_Float16)f3.z; a.h[15] = (_Float16)f3.w;

        const int koff = kb * 32 + half * 8;
        #pragma unroll
        for (int nt = 0; nt < 4; ++nt) {
            const int n = nt * 16 + l15;
            const v16h b = load_a_op(&W1t[n * FEAT + koff], &W1t[n * FEAT + koff + 16]);
            acc[nt] = __builtin_amdgcn_wmma_f32_16x16x32_f16(
                false, a.v, false, b, (short)0, acc[nt], false, false);
        }
    }

    // bias + tanh -> h1 (f16, row-major in LDS)
    #pragma unroll
    for (int nt = 0; nt < 4; ++nt) {
        const int col = nt * 16 + l15;
        const float bias = b1[col];
        #pragma unroll
        for (int r = 0; r < 8; ++r)
            h1[r + half * 8][col] = (_Float16)tanhf(acc[nt][r] + bias);
    }
    __syncthreads();

    // ---- GEMM2: h1[16 x 64] @ W2[64 x 64] ----
    v8f acc2[4] = {vz, vz, vz, vz};
    #pragma unroll
    for (int kb = 0; kb < 2; ++kb) {
        const int abase = kb * 32 + half * 8;
        const v16h a2 = load_a_op(&h1[l15][abase], &h1[l15][abase + 16]);
        const int koff = kb * 32 + half * 8;
        #pragma unroll
        for (int nt = 0; nt < 4; ++nt) {
            const int n = nt * 16 + l15;
            const v16h b = load_a_op(&W2t[n * HID + koff], &W2t[n * HID + koff + 16]);
            acc2[nt] = __builtin_amdgcn_wmma_f32_16x16x32_f16(
                false, a2, false, b, (short)0, acc2[nt], false, false);
        }
    }

    // bias + tanh -> h2 (f32 in LDS)
    #pragma unroll
    for (int nt = 0; nt < 4; ++nt) {
        const int col = nt * 16 + l15;
        const float bias = b2[col];
        #pragma unroll
        for (int r = 0; r < 8; ++r)
            h2[r + half * 8][col] = tanhf(acc2[nt][r] + bias);
    }
    __syncthreads();

    // ---- layer 3 (64 -> 1) + tile-level sum ----
    float e = 0.0f;
    if (lane < 16) {
        float s = b3[0];
        #pragma unroll 8
        for (int c = 0; c < HID; ++c)
            s += h2[lane][c] * W3[c];
        e = s;
    }
    #pragma unroll
    for (int o = 16; o > 0; o >>= 1)
        e += __shfl_xor(e, o, 32);
    if (lane == 0)
        atomicAdd(out, e);
}

// ---------------------------------------------------------------------------
extern "C" void kernel_launch(void* const* d_in, const int* in_sizes, int n_in,
                              void* d_out, int out_size, void* d_ws, size_t ws_size,
                              hipStream_t stream)
{
    (void)in_sizes; (void)n_in; (void)out_size; (void)ws_size;

    const float* pos   = (const float*)d_in[0];
    const float* spe   = (const float*)d_in[1];
    const float* theta = (const float*)d_in[2];
    const float* kn    = (const float*)d_in[3];
    const float* W1    = (const float*)d_in[4];
    const float* b1    = (const float*)d_in[5];
    const float* W2    = (const float*)d_in[6];
    const float* b2    = (const float*)d_in[7];
    const float* W3    = (const float*)d_in[8];
    const float* b3    = (const float*)d_in[9];
    const int*   nbr   = (const int*)d_in[10];
    float*       out   = (float*)d_out;

    char* ws = (char*)d_ws;
    float*    desc = (float*)ws;                            // 10000*576*4 = 23,040,000 B
    _Float16* W1t  = (_Float16*)(ws + 23040000);            //   576*64*2  =     73,728 B
    _Float16* W2t  = (_Float16*)(ws + 23040000 + 73728);    //    64*64*2  =      8,192 B

    prep_kernel<<<(FEAT * HID + 255) / 256, 256, 0, stream>>>(W1, W2, W1t, W2t, out);
    desc_kernel<<<NAT, 32, 0, stream>>>(pos, spe, theta, kn, nbr, desc);
    mlp_kernel<<<NAT / 16, 32, 0, stream>>>(desc, W1t, b1, W2t, b2, W3, b3, out);
}
